// ChildSumTreeLSTM_63848983823146
// MI455X (gfx1250) — matrix operand order repaired and checked by reading, not compiled, not run on hardware
//
#include <hip/hip_runtime.h>
#include <cstddef>

// ---------- CDNA5 WMMA types ----------
typedef __attribute__((ext_vector_type(16))) _Float16 v16h;
typedef __attribute__((ext_vector_type(8)))  _Float16 v8h;
typedef __attribute__((ext_vector_type(4)))  _Float16 v4h;
typedef __attribute__((ext_vector_type(8)))  float    v8f;

union Frag16 { v16h v; v8h h[2]; };

// Tree constants (complete 4-ary tree, depth 7)
#define TL_H      512
#define TL_NIOU   1536
#define TL_NODES  21845

// ---------------------------------------------------------------------------
__global__ void tl_zero(float* __restrict__ p, int n4) {
    int i = blockIdx.x * blockDim.x + threadIdx.x;
    if (i < n4) reinterpret_cast<float4*>(p)[i] = make_float4(0.f, 0.f, 0.f, 0.f);
}

// fp32 -> fp16 bulk convert (n4 = count of float4 groups)
__global__ void tl_cvt16(const float* __restrict__ src, _Float16* __restrict__ dst,
                         int n4) {
    int i = blockIdx.x * blockDim.x + threadIdx.x;
    if (i >= n4) return;
    float4 v = reinterpret_cast<const float4*>(src)[i];
    v4h h = { (_Float16)v.x, (_Float16)v.y, (_Float16)v.z, (_Float16)v.w };
    reinterpret_cast<v4h*>(dst)[i] = h;
}

// ---------------------------------------------------------------------------
// h_sum[m,:] = sum_{j=0..3} h[4*(s+m)+1+j, :]  (f32 in, f16 out for the GEMM)
// ---------------------------------------------------------------------------
__global__ void tl_gather(const float* __restrict__ h, _Float16* __restrict__ hsum16,
                          int s, int nl) {
    int idx = blockIdx.x * blockDim.x + threadIdx.x;   // float4 index
    if (idx >= nl * (TL_H / 4)) return;
    int m = idx >> 7;            // TL_H/4 == 128
    int q = (idx & 127) << 2;    // column
    size_t base = ((size_t)(4 * (s + m) + 1)) * TL_H + q;
    float4 a = *(const float4*)(h + base);
    float4 b = *(const float4*)(h + base + TL_H);
    float4 c = *(const float4*)(h + base + 2 * TL_H);
    float4 d = *(const float4*)(h + base + 3 * TL_H);
    v4h r = { (_Float16)(a.x + b.x + c.x + d.x), (_Float16)(a.y + b.y + c.y + d.y),
              (_Float16)(a.z + b.z + c.z + d.z), (_Float16)(a.w + b.w + c.w + d.w) };
    *(v4h*)(hsum16 + (size_t)m * TL_H + q) = r;
}

// ---------------------------------------------------------------------------
// Fused concat-K GEMM (all operands pre-converted f16):
//   out[m,n] = sum_k [A1(row m>>a1_log) | A2(row m)][k] * [B1 | B2][n,k] + bias[n]
// K = 1024. Block: 256 threads (8 waves), tile 128(M) x 128(N).
// K staged 64 at a time through double-buffered LDS. Per stage (rolled loop):
//   commit(cur) -> barrier -> issue(next) -> 16x WMMA from LDS
// Staged data lives in 8 NAMED uint4 registers (no arrays, no lambdas) so
// nothing can fall back to a stack alloca / scratch spill. Slot i is exactly
// 32 rows above slot 0, so B-loads and LDS stores use immediate offsets.
// ---------------------------------------------------------------------------
__global__ __launch_bounds__(256, 1)
void tl_gemm(const _Float16* __restrict__ A1, const _Float16* __restrict__ A2,
             const _Float16* __restrict__ B1, const _Float16* __restrict__ B2,
             const float* __restrict__ bias, float* __restrict__ out,
             int M, int Nout, int a1_log) {
    __shared__ _Float16 As[2][128 * 64];
    __shared__ _Float16 Bs[2][128 * 64];

    const int t    = threadIdx.x;
    const int bm   = blockIdx.y;
    const int bn   = blockIdx.x;
    const int wave = t >> 5;
    const int lane = t & 31;
    const int half = lane >> 4;
    const int l16  = lane & 15;
    const int wm   = wave >> 1;   // 0..3 -> 32-row strip
    const int wn   = wave & 1;    // 0..1 -> 64-col strip
    const int Mm1  = M - 1;

    // Staging geometry: slot i covers row (t>>3)+32*i, k-chunk (t&7)*8.
    const int row0 = t >> 3;                 // 0..31
    const int k8   = (t & 7) << 3;           // 0..56
    // A offsets must honor the row clamp (edge tiles) -> per-slot values.
    unsigned offA1[4], offA2[4];
#pragma unroll
    for (int i = 0; i < 4; ++i) {
        int am = bm * 128 + row0 + 32 * i;
        if (am > Mm1) am = Mm1;              // clamp (outputs masked in epilogue)
        offA1[i] = (unsigned)(((am >> a1_log) << 9) + k8);
        offA2[i] = (unsigned)((am << 9) + k8);
    }
    // B rows are always in range: affine in i (+32 rows = +16384 halves).
    const unsigned offB0 = (unsigned)(((bn * 128 + row0) << 9) + k8);
    const unsigned offL0 = (unsigned)(row0 * 64 + k8);   // +2048 halves per slot

    v8f acc[2][4] = {};
    uint4 ra0, ra1, ra2, ra3, rb0, rb1, rb2, rb3;   // in-flight staged chunks

#define TL_ISSUE(ks_)                                                         \
    {                                                                         \
        const bool p1 = ((ks_) < 8);                                          \
        const int  kb = p1 ? (ks_) * 64 : (ks_) * 64 - 512;                   \
        const _Float16* Ab = (p1 ? A1 : A2) + kb;                             \
        const _Float16* Bb = (p1 ? B1 : B2) + kb + offB0;                     \
        ra0 = *(const uint4*)(Ab + (p1 ? offA1[0] : offA2[0]));               \
        ra1 = *(const uint4*)(Ab + (p1 ? offA1[1] : offA2[1]));               \
        ra2 = *(const uint4*)(Ab + (p1 ? offA1[2] : offA2[2]));               \
        ra3 = *(const uint4*)(Ab + (p1 ? offA1[3] : offA2[3]));               \
        rb0 = *(const uint4*)(Bb);                                            \
        rb1 = *(const uint4*)(Bb + 16384);                                    \
        rb2 = *(const uint4*)(Bb + 32768);                                    \
        rb3 = *(const uint4*)(Bb + 49152);                                    \
    }

#define TL_COMMIT(buf_)                                                       \
    {                                                                         \
        _Float16* Ad = As[buf_] + offL0;                                      \
        _Float16* Bd = Bs[buf_] + offL0;                                      \
        *(uint4*)(Ad)        = ra0;  *(uint4*)(Bd)        = rb0;              \
        *(uint4*)(Ad + 2048) = ra1;  *(uint4*)(Bd + 2048) = rb1;              \
        *(uint4*)(Ad + 4096) = ra2;  *(uint4*)(Bd + 4096) = rb2;              \
        *(uint4*)(Ad + 6144) = ra3;  *(uint4*)(Bd + 6144) = rb3;              \
    }

    TL_ISSUE(0)

#pragma unroll 1
    for (int ks = 0; ks < 16; ++ks) {   // 16 * 64 = K 1024 (keep rolled!)
        const int cur = ks & 1;
        TL_COMMIT(cur)                  // waits for stage ks loads, LDS store
        __syncthreads();
        if (ks < 15) TL_ISSUE(ks + 1)   // next stage's loads fly during WMMAs

        const _Float16* __restrict__ Ac = As[cur];
        const _Float16* __restrict__ Bc = Bs[cur];
#pragma unroll
        for (int kk = 0; kk < 64; kk += 32) {
            Frag16 af[2], bf[4];
#pragma unroll
            for (int sm = 0; sm < 2; ++sm) {
                int r = 32 * wm + 16 * sm + l16;
                af[sm].h[0] = *(const v8h*)(Ac + r * 64 + kk + 8 * half);
                af[sm].h[1] = *(const v8h*)(Ac + r * 64 + kk + 16 + 8 * half);
            }
#pragma unroll
            for (int sn = 0; sn < 4; ++sn) {
                int c = 64 * wn + 16 * sn + l16;
                bf[sn].h[0] = *(const v8h*)(Bc + c * 64 + kk + 8 * half);
                bf[sn].h[1] = *(const v8h*)(Bc + c * 64 + kk + 16 + 8 * half);
            }
#pragma unroll
            for (int sm = 0; sm < 2; ++sm)
#pragma unroll
                for (int sn = 0; sn < 4; ++sn)
                    acc[sm][sn] = __builtin_amdgcn_wmma_f32_16x16x32_f16(
                        false, af[sm].v, false, bf[sn].v,
                        (short)0, acc[sm][sn], false, false);
        }
    }
#undef TL_ISSUE
#undef TL_COMMIT

    // ---- epilogue: add bias, row-guarded f32 stores ----
#pragma unroll
    for (int sm = 0; sm < 2; ++sm) {
#pragma unroll
        for (int sn = 0; sn < 4; ++sn) {
            int gc  = bn * 128 + 64 * wn + 16 * sn + l16;
            float bv = bias[gc];
            int gm0 = bm * 128 + 32 * wm + 16 * sm + 8 * half;
#pragma unroll
            for (int v = 0; v < 8; ++v) {
                int gm = gm0 + v;
                if (gm < M)
                    out[(size_t)gm * Nout + gc] = acc[sm][sn][v] + bv;
            }
        }
    }
}

// ---------------------------------------------------------------------------
// Gates + cell update (reference applies NO sigmoid to f — kept as-is).
// Also maintains the f16 mirror of h used as GEMM input at the next level.
// ---------------------------------------------------------------------------
__global__ void tl_combine(const float* __restrict__ iou,
                           const float* __restrict__ f,
                           float* __restrict__ h, float* __restrict__ c,
                           _Float16* __restrict__ h16,
                           int s, int nl) {
    int idx = blockIdx.x * blockDim.x + threadIdx.x;
    if (idx >= nl * TL_H) return;
    int m   = idx >> 9;          // TL_H == 512
    int col = idx & 511;
    int n   = s + m;

    float iv = iou[(size_t)m * TL_NIOU + col];
    float ov = iou[(size_t)m * TL_NIOU + 512 + col];
    float uv = iou[(size_t)m * TL_NIOU + 1024 + col];
    float i_ = 1.f / (1.f + __expf(-iv));
    float o_ = 1.f / (1.f + __expf(-ov));
    float u_ = tanhf(uv);

    float fc = 0.f;
#pragma unroll
    for (int j = 0; j < 4; ++j) {
        int cr = 4 * n + 1 + j;
        fc += f[(size_t)(4 * m + j) * TL_H + col] * c[(size_t)cr * TL_H + col];
    }
    float cn = i_ * u_ + fc;
    float hv = o_ * tanhf(cn);
    c[(size_t)n * TL_H + col]   = cn;
    h[(size_t)n * TL_H + col]   = hv;
    h16[(size_t)n * TL_H + col] = (_Float16)hv;
}

// ---------------------------------------------------------------------------
extern "C" void kernel_launch(void* const* d_in, const int* in_sizes, int n_in,
                              void* d_out, int out_size, void* d_ws, size_t ws_size,
                              hipStream_t stream) {
    const float* x     = (const float*)d_in[0];
    // d_in[1] = children (implicit: child(n,j) = 4n+1+j for a complete tree)
    const float* W_iou = (const float*)d_in[2];
    const float* b_iou = (const float*)d_in[3];
    const float* W_f   = (const float*)d_in[4];
    const float* b_f   = (const float*)d_in[5];
    const float* U_iou = (const float*)d_in[6];
    const float* U_f   = (const float*)d_in[7];

    float* h_out = (float*)d_out;                              // [N, 512]
    float* c_out = h_out + (size_t)TL_NODES * TL_H;            // [N, 512]

    // Workspace layout
    char* ws = (char*)d_ws;
    float*    iou_buf = (float*)ws;                       ws += (size_t)4096 * TL_NIOU * 4;
    float*    f_buf   = (float*)ws;                       ws += (size_t)16384 * TL_H * 4;
    _Float16* hsum16  = (_Float16*)ws;                    ws += (size_t)4096 * TL_H * 2;
    _Float16* x16     = (_Float16*)ws;                    ws += (size_t)TL_NODES * TL_H * 2;
    _Float16* h16     = (_Float16*)ws;                    ws += (size_t)TL_NODES * TL_H * 2;
    _Float16* W_iou16 = (_Float16*)ws;                    ws += (size_t)TL_NIOU * TL_H * 2;
    _Float16* U_iou16 = (_Float16*)ws;                    ws += (size_t)TL_NIOU * TL_H * 2;
    _Float16* W_f16   = (_Float16*)ws;                    ws += (size_t)TL_H * TL_H * 2;
    _Float16* U_f16   = (_Float16*)ws;

    // 1) zero h/c outputs and the f16 leaf mirror
    {
        int n4 = 2 * TL_NODES * TL_H / 4;
        tl_zero<<<(n4 + 255) / 256, 256, 0, stream>>>(h_out, n4);
        int nh4 = TL_NODES * TL_H * 2 / 16;   // h16 bytes / 16
        tl_zero<<<(nh4 + 255) / 256, 256, 0, stream>>>((float*)h16, nh4);
    }
    // 2) one-time f16 conversions (x + all weights)
    {
        int nx = TL_NODES * TL_H / 4;
        tl_cvt16<<<(nx + 255) / 256, 256, 0, stream>>>(x, x16, nx);
        int nw = TL_NIOU * TL_H / 4;
        tl_cvt16<<<(nw + 255) / 256, 256, 0, stream>>>(W_iou, W_iou16, nw);
        tl_cvt16<<<(nw + 255) / 256, 256, 0, stream>>>(U_iou, U_iou16, nw);
        int nf = TL_H * TL_H / 4;
        tl_cvt16<<<(nf + 255) / 256, 256, 0, stream>>>(W_f, W_f16, nf);
        tl_cvt16<<<(nf + 255) / 256, 256, 0, stream>>>(U_f, U_f16, nf);
    }

    // offs[l] = (4^l - 1)/3 : node offset of level l
    static const int offs[9] = {0, 1, 5, 21, 85, 341, 1365, 5461, 21845};

    // 3) bottom-up level sweep
    for (int l = 6; l >= 0; --l) {
        int s  = offs[l];
        int nl = offs[l + 1] - offs[l];

        // h_sum over the 4 children of each node (f16 out)
        {
            int n4 = nl * (TL_H / 4);
            tl_gather<<<(n4 + 255) / 256, 256, 0, stream>>>(h_out, hsum16, s, nl);
        }
        // iou = [x | h_sum] @ [W_iou | U_iou]^T + b_iou   (M=nl, N=1536, K=1024)
        {
            dim3 g(TL_NIOU / 128, (nl + 127) / 128);
            tl_gemm<<<g, 256, 0, stream>>>(x16 + (size_t)s * TL_H, hsum16,
                                           W_iou16, U_iou16, b_iou, iou_buf,
                                           nl, TL_NIOU, /*a1_log=*/0);
        }
        // f = [x(bcast over 4 children) | h_child] @ [W_f | U_f]^T + b_f
        //     (M=4*nl, N=512, K=1024); child rows of h are contiguous at 4s+1
        {
            dim3 g(TL_H / 128, (4 * nl + 127) / 128);
            tl_gemm<<<g, 256, 0, stream>>>(x16 + (size_t)s * TL_H,
                                           h16 + (size_t)(4 * s + 1) * TL_H,
                                           W_f16, U_f16, b_f, f_buf,
                                           4 * nl, TL_H, /*a1_log=*/2);
        }
        // gates + cell/hidden update for this level
        {
            int ne = nl * TL_H;
            tl_combine<<<(ne + 255) / 256, 256, 0, stream>>>(iou_buf, f_buf,
                                                             h_out, c_out, h16,
                                                             s, nl);
        }
    }
}
